// MobileMQA_11450382811328
// MI455X (gfx1250) — matrix operand order repaired and checked by reading, not compiled, not run on hardware
//
#include <hip/hip_runtime.h>
#include <hip/hip_bf16.h>
#include <math.h>
#include <stdint.h>

// ---- problem constants (match reference) ----
#define NB     8
#define CCH    256
#define HHH    64
#define WWW    64
#define NNQ    4096   // H*W
#define MMK    1024   // (H/2)*(W/2)
#define HD     64
#define NHEADS 4
#define EPSV   1e-5f

typedef _Float16 v16h __attribute__((ext_vector_type(16)));
typedef _Float16 v8h  __attribute__((ext_vector_type(8)));
typedef float    v8f  __attribute__((ext_vector_type(8)));

// ---------------- fragment load helpers (CDNA5 WMMA layouts) ----------------
// A-matrix 16x32 f16: lane = row(0..15) in lane%16; halves 0..7 = K k0..k0+7,
// halves 8..15 = K k0+16..k0+23, with k0 = 8*(lane>=16).
__device__ static inline v16h load_a16(const _Float16* base, int ld) {
    int lane = threadIdx.x & 31;
    int row  = lane & 15;
    int k0   = (lane >> 4) << 3;
    const _Float16* p = base + row * ld + k0;
    v8h lo = *(const v8h*)(p);
    v8h hi = *(const v8h*)(p + 16);
    v16h r;
#pragma unroll
    for (int i = 0; i < 8; ++i) { r[i] = lo[i]; r[i + 8] = hi[i]; }
    return r;
}

// B-matrix 32x16 f16 (stored [col][k], k contiguous): lane column = lane%16,
// halves 0..7 = K k0..k0+7, halves 8..15 = K k0+16..k0+23, k0 = 8*(lane>=16).
__device__ static inline v16h load_b16(const _Float16* base, int ld) {
    int lane = threadIdx.x & 31;
    int col  = lane & 15;
    int k0   = (lane >> 4) << 3;
    const _Float16* p = base + col * ld + k0;
    v8h lo = *(const v8h*)(p);
    v8h hi = *(const v8h*)(p + 16);
    v16h r;
#pragma unroll
    for (int i = 0; i < 8; ++i) { r[i] = lo[i]; r[i + 8] = hi[i]; }
    return r;
}

__device__ static inline v8f wmma_f16(v16h a, v16h b, v8f c) {
    return __builtin_amdgcn_wmma_f32_16x16x32_f16(false, a, false, b,
                                                  (short)0, c, false, false);
}

// CDNA5 async global->LDS copy, 16 bytes per lane (tracked by ASYNCcnt).
// VDST = VGPR holding LDS byte address; VADDR = 64-bit global address.
__device__ static inline void async_copy16(uint32_t lds_addr, const void* gaddr) {
    uint64_t ga = (uint64_t)(uintptr_t)gaddr;
    asm volatile("global_load_async_to_lds_b128 %0, %1, off"
                 :: "v"(lds_addr), "v"(ga) : "memory");
}
__device__ static inline void wait_async0() {
    asm volatile("s_wait_asynccnt 0x0" ::: "memory");
}
__device__ static inline uint32_t lds_lo32(const void* p) {
    return (uint32_t)(uintptr_t)p;
}

// ---------------- 0) f32 -> f16 weight conversion ----------------
__global__ void k_f32_to_f16(const float* __restrict__ src,
                             _Float16* __restrict__ dst, int n) {
    int i = blockIdx.x * blockDim.x + threadIdx.x;
    if (i < n) dst[i] = (_Float16)src[i];
}

// ---------------- 1) GroupNorm(1 group) statistics ----------------
__global__ __launch_bounds__(256)
void k_gn_stats(const float* __restrict__ x, float* __restrict__ stats) {
    const int b   = blockIdx.x;
    const int tid = threadIdx.x;
    const int CHW = CCH * NNQ;
    const float* xb = x + (size_t)b * CHW;
    float s1 = 0.f, s2 = 0.f;
    for (int i = tid; i < CHW; i += 256) {
        float v = xb[i];
        s1 += v; s2 += v * v;
    }
    __shared__ float ss1[256], ss2[256];
    ss1[tid] = s1; ss2[tid] = s2;
    __syncthreads();
    for (int off = 128; off > 0; off >>= 1) {
        if (tid < off) { ss1[tid] += ss1[tid + off]; ss2[tid] += ss2[tid + off]; }
        __syncthreads();
    }
    if (tid == 0) {
        float mu  = ss1[0] / (float)CHW;
        float var = ss2[0] / (float)CHW - mu * mu;
        stats[b * 2 + 0] = mu;
        stats[b * 2 + 1] = rsqrtf(var + EPSV);
    }
}

// ---- 2) normalize + affine with LDS tile transpose; write xn_t[B][N][C] fp16 ----
// grid = (N/16, C/16, B), block = 256 (16x16 tile); both sides coalesced.
__global__ __launch_bounds__(256)
void k_normalize(const float* __restrict__ x, const float* __restrict__ stats,
                 const float* __restrict__ gw, const float* __restrict__ gb,
                 _Float16* __restrict__ xn_t) {
    __shared__ float tile[16][17];
    const int n0 = blockIdx.x * 16, c0 = blockIdx.y * 16, b = blockIdx.z;
    const int tx = threadIdx.x & 15, ty = threadIdx.x >> 4;
    const float mu = stats[b * 2 + 0], rs = stats[b * 2 + 1];
    // read: channel row c0+ty, spatial col n0+tx (contiguous in n)
    float v = x[((size_t)b * CCH + c0 + ty) * NNQ + n0 + tx];
    tile[ty][tx] = (v - mu) * rs;
    __syncthreads();
    // write: spatial row n0+ty, channel col c0+tx (contiguous in c)
    const int c = c0 + tx;
    float o = tile[tx][ty] * gw[c] + gb[c];
    xn_t[((size_t)b * NNQ + n0 + ty) * CCH + c] = (_Float16)o;
}

// ---------------- 3) 2x2 avg pool -> xds_t[B][M][C] fp16 ----------------
__global__ __launch_bounds__(256)
void k_pool(const _Float16* __restrict__ xn_t, _Float16* __restrict__ xds_t) {
    const int m = blockIdx.x, b = blockIdx.y, c = threadIdx.x;
    const int hy = m >> 5, wx = m & 31;
    const int n0 = (hy * 2) * WWW + wx * 2;
    const _Float16* base = xn_t + (size_t)b * NNQ * CCH;
    float s = (float)base[(size_t)(n0      ) * CCH + c]
            + (float)base[(size_t)(n0 + 1  ) * CCH + c]
            + (float)base[(size_t)(n0 + WWW) * CCH + c]
            + (float)base[(size_t)(n0 + WWW + 1) * CCH + c];
    xds_t[((size_t)b * MMK + m) * CCH + c] = (_Float16)(s * 0.25f);
}

// ---------------- 4) generic f16 GEMM: Out[R][Co] = A[R][K] * W[Co][K]^T ----------------
// grid = (R/16, Co/64, B), block = 128 (4 waves, one 16x16 col-tile each)
__global__ __launch_bounds__(128)
void k_gemm_f16(const _Float16* __restrict__ A, const _Float16* __restrict__ W,
                _Float16* __restrict__ Out, int K, int Co, int R, int transposed,
                long aStride, long oStride) {
    const int w     = threadIdx.x >> 5;
    const int lane  = threadIdx.x & 31;
    const int rtile = blockIdx.x * 16;
    const int cbase = blockIdx.y * 64 + w * 16;
    const int b     = blockIdx.z;

    const _Float16* aRow = A + (size_t)b * aStride + (size_t)rtile * K;
    const _Float16* wRow = W + (size_t)cbase * K;

    v8f acc = {0.f, 0.f, 0.f, 0.f, 0.f, 0.f, 0.f, 0.f};
    for (int k = 0; k < K; k += 32) {
        __builtin_prefetch(aRow + k + 32, 0, 3);
        __builtin_prefetch(wRow + k + 32, 0, 3);
        v16h af = load_a16(aRow + k, K);
        v16h bf = load_b16(wRow + k, K);
        acc = wmma_f16(af, bf, acc);
    }

    const int col   = lane & 15;
    const int rbase = (lane >> 4) * 8;
    _Float16* ob = Out + (size_t)b * oStride;
#pragma unroll
    for (int r = 0; r < 8; ++r) {
        int row = rtile + r + rbase;
        if (transposed) ob[(size_t)(cbase + col) * R + row] = (_Float16)acc[r];
        else            ob[(size_t)row * Co + cbase + col]  = (_Float16)acc[r];
    }
}

// ---------------- 5) flash attention (multi-query, M=1024) ----------------
// grid = (N/64, NHEADS, B), block = 128; each wave: 16 query rows x full hd=64.
// K/V tiles are shared by all 4 waves (multi-query) -> stage them in LDS with
// double-buffered CDNA5 async global->LDS DMA overlapped with WMMA compute.
__global__ __launch_bounds__(128)
void k_attention(const _Float16* __restrict__ q_t, const _Float16* __restrict__ k_t,
                 const _Float16* __restrict__ v_t, _Float16* __restrict__ ao_t) {
    const int tid    = threadIdx.x;
    const int w      = tid >> 5;
    const int lane   = tid & 31;
    const int h      = blockIdx.y;
    const int b      = blockIdx.z;
    const int n_base = blockIdx.x * 64 + w * 16;
    const float scale = 0.125f; // hd^-0.5

    const _Float16* qb = q_t + (size_t)b * NNQ * CCH + (size_t)n_base * CCH + h * HD;
    const _Float16* kb = k_t + (size_t)b * MMK * HD;
    const _Float16* vb = v_t + (size_t)b * HD * MMK;

    // double-buffered K tile [32 m][64 d] and V tile [64 d][32 m]
    __shared__ __align__(16) _Float16 ldsK[2][32 * 64];
    __shared__ __align__(16) _Float16 ldsV[2][64 * 32];
    __shared__ __align__(16) _Float16 pbuf[4][16 * 32];

    // cooperative async stage of one 32-wide K/V tile into LDS buffer `buf`
    auto stage = [&](int buf, int m0) {
        // K tile: 32 rows x 128B, fully contiguous (4 KB): 2 x 16B per thread
        const char* gk = (const char*)(kb + (size_t)m0 * HD);
        uint32_t lk = lds_lo32(&ldsK[buf][0]);
        async_copy16(lk + tid * 16,         gk + tid * 16);
        async_copy16(lk + (tid + 128) * 16, gk + (tid + 128) * 16);
        // V tile: 64 rows x 64B, row stride = M halves: 2 x 16B per thread
        uint32_t lv = lds_lo32(&ldsV[buf][0]);
        {
            int row = tid >> 2, c4 = tid & 3;
            const char* gv = (const char*)(vb + (size_t)row * MMK + m0);
            async_copy16(lv + row * 64 + c4 * 16, gv + c4 * 16);
        }
        {
            int t = tid + 128;
            int row = t >> 2, c4 = t & 3;
            const char* gv = (const char*)(vb + (size_t)row * MMK + m0);
            async_copy16(lv + row * 64 + c4 * 16, gv + c4 * 16);
        }
    };

    // Q fragments (K = d, two chunks of 32)
    v16h qa0 = load_a16(qb + 0,  CCH);
    v16h qa1 = load_a16(qb + 32, CCH);

    v8f acc0 = {0,0,0,0,0,0,0,0}, acc1 = acc0, acc2 = acc0, acc3 = acc0;
    float rowmax[8], rowsum[8];
#pragma unroll
    for (int r = 0; r < 8; ++r) { rowmax[r] = -1e30f; rowsum[r] = 0.f; }

    _Float16* pb = &pbuf[w][0];
    const int col   = lane & 15;
    const int rbase = (lane >> 4) * 8;

    stage(0, 0);  // prologue
    for (int m0 = 0; m0 < MMK; m0 += 32) {
        const int buf = (m0 >> 5) & 1;
        wait_async0();        // our async copies (incl. current buf) landed
        __syncthreads();      // whole block: current buf ready, prev compute done
        if (m0 + 32 < MMK) stage(buf ^ 1, m0 + 32);  // overlap next DMA

        const _Float16* Kb = &ldsK[buf][0];
        const _Float16* Vb = &ldsV[buf][0];

        // scores: two 16x16 m-tiles, K over d=64 (B-fragments from LDS)
        v8f s0 = {0,0,0,0,0,0,0,0}, s1 = s0;
        s0 = wmma_f16(qa0, load_b16(Kb + 0,            64), s0);
        s0 = wmma_f16(qa1, load_b16(Kb + 32,           64), s0);
        s1 = wmma_f16(qa0, load_b16(Kb + 16 * 64 + 0,  64), s1);
        s1 = wmma_f16(qa1, load_b16(Kb + 16 * 64 + 32, 64), s1);

        // online softmax (row reductions across the 16 lanes of each half)
#pragma unroll
        for (int r = 0; r < 8; ++r) {
            float sv0 = s0[r] * scale;
            float sv1 = s1[r] * scale;
            float mnew = fmaxf(fmaxf(sv0, sv1), rowmax[r]);
            mnew = fmaxf(mnew, __shfl_xor(mnew, 1, 32));
            mnew = fmaxf(mnew, __shfl_xor(mnew, 2, 32));
            mnew = fmaxf(mnew, __shfl_xor(mnew, 4, 32));
            mnew = fmaxf(mnew, __shfl_xor(mnew, 8, 32));
            float alpha = expf(rowmax[r] - mnew);
            float p0 = expf(sv0 - mnew);
            float p1 = expf(sv1 - mnew);
            float ps = p0 + p1;
            ps += __shfl_xor(ps, 1, 32);
            ps += __shfl_xor(ps, 2, 32);
            ps += __shfl_xor(ps, 4, 32);
            ps += __shfl_xor(ps, 8, 32);
            rowsum[r] = rowsum[r] * alpha + ps;
            rowmax[r] = mnew;
            acc0[r] *= alpha; acc1[r] *= alpha; acc2[r] *= alpha; acc3[r] *= alpha;
            pb[(r + rbase) * 32 + col]      = (_Float16)p0;
            pb[(r + rbase) * 32 + col + 16] = (_Float16)p1;
        }
        __syncthreads();  // make P visible for the cross-lane A-fragment reload

        v16h pa = load_a16(pb, 32);
        acc0 = wmma_f16(pa, load_b16(Vb + 0 * 16 * 32, 32), acc0);
        acc1 = wmma_f16(pa, load_b16(Vb + 1 * 16 * 32, 32), acc1);
        acc2 = wmma_f16(pa, load_b16(Vb + 2 * 16 * 32, 32), acc2);
        acc3 = wmma_f16(pa, load_b16(Vb + 3 * 16 * 32, 32), acc3);
        // top-of-loop barrier protects pbuf / ldsK / ldsV reuse
    }

    _Float16* ob = ao_t + (size_t)b * NNQ * CCH + h * HD;
#pragma unroll
    for (int r = 0; r < 8; ++r) {
        float inv = 1.0f / rowsum[r];
        int n = n_base + r + rbase;
        ob[(size_t)n * CCH + 0 * 16 + col] = (_Float16)(acc0[r] * inv);
        ob[(size_t)n * CCH + 1 * 16 + col] = (_Float16)(acc1[r] * inv);
        ob[(size_t)n * CCH + 2 * 16 + col] = (_Float16)(acc2[r] * inv);
        ob[(size_t)n * CCH + 3 * 16 + col] = (_Float16)(acc3[r] * inv);
    }
}

// ---------------- 6) output projection + gamma-scaled residual ----------------
// grid = (N/16, C/64, B), block = 128
__global__ __launch_bounds__(128)
void k_outproj(const _Float16* __restrict__ ao_t, const _Float16* __restrict__ wo_h,
               const float* __restrict__ x, const float* __restrict__ gamma,
               float* __restrict__ y) {
    const int w     = threadIdx.x >> 5;
    const int lane  = threadIdx.x & 31;
    const int rtile = blockIdx.x * 16;
    const int cbase = blockIdx.y * 64 + w * 16;
    const int b     = blockIdx.z;

    const _Float16* aRow = ao_t + (size_t)b * NNQ * CCH + (size_t)rtile * CCH;
    const _Float16* wRow = wo_h + (size_t)cbase * CCH;

    v8f acc = {0.f, 0.f, 0.f, 0.f, 0.f, 0.f, 0.f, 0.f};
    for (int k = 0; k < CCH; k += 32) {
        __builtin_prefetch(aRow + k + 32, 0, 3);
        __builtin_prefetch(wRow + k + 32, 0, 3);
        v16h af = load_a16(aRow + k, CCH);
        v16h bf = load_b16(wRow + k, CCH);
        acc = wmma_f16(af, bf, acc);
    }

    const int col   = lane & 15;
    const int rbase = (lane >> 4) * 8;
    const int o     = cbase + col;
    const float g   = gamma[o];
#pragma unroll
    for (int r = 0; r < 8; ++r) {
        int n = rtile + r + rbase;
        size_t idx = ((size_t)b * CCH + o) * NNQ + n;
        y[idx] = x[idx] + g * acc[r];
    }
}

// ---------------- host ----------------
extern "C" void kernel_launch(void* const* d_in, const int* in_sizes, int n_in,
                              void* d_out, int out_size, void* d_ws, size_t ws_size,
                              hipStream_t stream) {
    (void)in_sizes; (void)n_in; (void)out_size; (void)ws_size;
    const float* x     = (const float*)d_in[0];
    const float* gn_w  = (const float*)d_in[1];
    const float* gn_b  = (const float*)d_in[2];
    const float* wq    = (const float*)d_in[3];
    const float* wk    = (const float*)d_in[4];
    const float* wv    = (const float*)d_in[5];
    const float* wo    = (const float*)d_in[6];
    const float* gamma = (const float*)d_in[7];
    float* out = (float*)d_out;

    char* ws = (char*)d_ws;
    size_t off = 0;
    auto alloc = [&](size_t bytes) {
        size_t cur = off;
        off = (off + bytes + 255) & ~(size_t)255;
        return cur;
    };
    float*    stats = (float*)   (ws + alloc(NB * 2 * sizeof(float)));
    _Float16* wq_h  = (_Float16*)(ws + alloc((size_t)CCH * CCH * 2));
    _Float16* wk_h  = (_Float16*)(ws + alloc((size_t)HD  * CCH * 2));
    _Float16* wv_h  = (_Float16*)(ws + alloc((size_t)HD  * CCH * 2));
    _Float16* wo_h  = (_Float16*)(ws + alloc((size_t)CCH * CCH * 2));
    _Float16* xn_t  = (_Float16*)(ws + alloc((size_t)NB * NNQ * CCH * 2));
    _Float16* xds_t = (_Float16*)(ws + alloc((size_t)NB * MMK * CCH * 2));
    _Float16* q_t   = (_Float16*)(ws + alloc((size_t)NB * NNQ * CCH * 2));
    _Float16* k_t   = (_Float16*)(ws + alloc((size_t)NB * MMK * HD  * 2));
    _Float16* v_t   = (_Float16*)(ws + alloc((size_t)NB * HD  * MMK * 2));
    _Float16* ao_t  = (_Float16*)(ws + alloc((size_t)NB * NNQ * CCH * 2));

    // weights -> fp16
    k_f32_to_f16<<<(CCH * CCH + 255) / 256, 256, 0, stream>>>(wq, wq_h, CCH * CCH);
    k_f32_to_f16<<<(HD * CCH + 255) / 256, 256, 0, stream>>>(wk, wk_h, HD * CCH);
    k_f32_to_f16<<<(HD * CCH + 255) / 256, 256, 0, stream>>>(wv, wv_h, HD * CCH);
    k_f32_to_f16<<<(CCH * CCH + 255) / 256, 256, 0, stream>>>(wo, wo_h, CCH * CCH);

    // GroupNorm
    k_gn_stats<<<NB, 256, 0, stream>>>(x, stats);
    k_normalize<<<dim3(NNQ / 16, CCH / 16, NB), 256, 0, stream>>>(x, stats, gn_w, gn_b, xn_t);
    k_pool<<<dim3(MMK, NB), 256, 0, stream>>>(xn_t, xds_t);

    // projections
    k_gemm_f16<<<dim3(NNQ / 16, CCH / 64, NB), 128, 0, stream>>>(
        xn_t, wq_h, q_t, CCH, CCH, NNQ, 0, (long)NNQ * CCH, (long)NNQ * CCH);
    k_gemm_f16<<<dim3(MMK / 16, 1, NB), 128, 0, stream>>>(
        xds_t, wk_h, k_t, CCH, HD, MMK, 0, (long)MMK * CCH, (long)MMK * HD);
    k_gemm_f16<<<dim3(MMK / 16, 1, NB), 128, 0, stream>>>(
        xds_t, wv_h, v_t, CCH, HD, MMK, 1, (long)MMK * CCH, (long)HD * MMK);

    // attention
    k_attention<<<dim3(NNQ / 64, NHEADS, NB), 128, 0, stream>>>(q_t, k_t, v_t, ao_t);

    // output projection + residual
    k_outproj<<<dim3(NNQ / 16, CCH / 64, NB), 128, 0, stream>>>(ao_t, wo_h, x, gamma, out);
}
